// MultiHeadSelfAttention_62173946577659
// MI455X (gfx1250) — matrix operand order repaired and checked by reading, not compiled, not run on hardware
//
#include <hip/hip_runtime.h>
#include <hip/hip_bf16.h>

// ---- problem constants (from reference) ----
#define BB   4
#define LLN  2048
#define EE   1024
#define HH   16
#define DD   64
#define AAH  1024   // H*D
#define N3A  3072   // 3*A

typedef __attribute__((ext_vector_type(16))) __bf16 v16bf;
typedef __attribute__((ext_vector_type(8)))  __bf16 v8bf;
typedef __attribute__((ext_vector_type(8)))  float  v8f;

__device__ __forceinline__ v8bf ld8(const __bf16* p) { return *(const v8bf*)p; }

__device__ __forceinline__ v16bf cat8(v8bf lo, v8bf hi) {
  return __builtin_shufflevector(lo, hi, 0,1,2,3,4,5,6,7,8,9,10,11,12,13,14,15);
}

__device__ __forceinline__ v8f wmma_bf16(v16bf a, v16bf b, v8f c) {
  // (neg_a, A, neg_b, B, c_mod, C, reuse_a, reuse_b)
  return __builtin_amdgcn_wmma_f32_16x16x32_bf16(false, a, false, b, (short)0, c,
                                                 false, false);
}

// ---------------- fp32 -> bf16 convert ----------------
__global__ void cvt_f32_bf16(const float* __restrict__ src,
                             __bf16* __restrict__ dst, int n) {
  int i = blockIdx.x * blockDim.x + threadIdx.x;
  int stride = gridDim.x * blockDim.x;
  for (; i < n; i += stride) dst[i] = (__bf16)src[i];
}

// ------------- fp32 W(KxN) -> bf16 WT(NxK), tiled 64x64 transpose -------------
__global__ __launch_bounds__(256) void cvt_transpose_bf16(
    const float* __restrict__ W, __bf16* __restrict__ WT, int K, int N) {
  __shared__ __bf16 tile[64][72];   // +8 pad: row stride 144B, still 16B aligned
  const int k0 = blockIdx.y * 64, n0 = blockIdx.x * 64;
  const int t = threadIdx.x;
  {
    int r = t >> 2, seg = t & 3;    // r: k row, seg: 16-wide n segment
    const float* src = W + (size_t)(k0 + r) * N + n0 + seg * 16;
    v8bf a, b;
#pragma unroll
    for (int j = 0; j < 8; ++j) { a[j] = (__bf16)src[j]; b[j] = (__bf16)src[8 + j]; }
    *(v8bf*)&tile[r][seg * 16]     = a;
    *(v8bf*)&tile[r][seg * 16 + 8] = b;
  }
  __syncthreads();
  {
    int rr = t >> 2, seg = t & 3;   // rr: n row, seg: 16-wide k segment
    v8bf a, b;
#pragma unroll
    for (int j = 0; j < 8; ++j) {
      a[j] = tile[seg * 16 + j][rr];
      b[j] = tile[seg * 16 + 8 + j][rr];
    }
    __bf16* dst = WT + (size_t)(n0 + rr) * K + k0 + seg * 16;
    *(v8bf*)dst       = a;
    *(v8bf*)(dst + 8) = b;
  }
}

// ---------------- 64x64-tile bf16 WMMA GEMM, B pre-transposed ----------------
// C = A(MxK) * B(KxN) with B supplied as BT(NxK), fp32 accumulate.
// mode 0: write fp32 C row-major into Cout
// mode 1: scatter QKV columns into q/k (B,H,L,D) and v (B,H,D,L) as bf16
__global__ __launch_bounds__(128) void gemm_bf16_64x64(
    const __bf16* __restrict__ A, const __bf16* __restrict__ BT,
    int M, int N, int K, int mode,
    float* __restrict__ Cout,
    __bf16* __restrict__ qo, __bf16* __restrict__ ko, __bf16* __restrict__ vto) {
  __shared__ __bf16 At[64 * 32];   // (m, k)
  __shared__ __bf16 Bt[64 * 32];   // (n, k)

  const int t    = threadIdx.x;
  const int lane = t & 31;
  const int wave = t >> 5;
  const int half = lane >> 4;
  const int lm   = lane & 15;
  const int m0   = blockIdx.y * 64;
  const int n0   = blockIdx.x * 64;

  v8f acc[4] = {};

  for (int k0 = 0; k0 < K; k0 += 32) {
    __syncthreads();
    {
      int r = t >> 1, seg = t & 1;
      const __bf16* asrc = A + (size_t)(m0 + r) * K + k0 + seg * 16;
      v8bf x0 = ld8(asrc), x1 = ld8(asrc + 8);
      *(v8bf*)(At + r * 32 + seg * 16)     = x0;
      *(v8bf*)(At + r * 32 + seg * 16 + 8) = x1;
      const __bf16* bsrc = BT + (size_t)(n0 + r) * K + k0 + seg * 16;
      v8bf y0 = ld8(bsrc), y1 = ld8(bsrc + 8);
      *(v8bf*)(Bt + r * 32 + seg * 16)     = y0;
      *(v8bf*)(Bt + r * 32 + seg * 16 + 8) = y1;
    }
    __syncthreads();

    int arow = wave * 16 + lm;
    int akb  = half ? 16 : 0;
    v16bf af = cat8(ld8(At + arow * 32 + akb), ld8(At + arow * 32 + akb + 8));
    v16bf bf4[4];
#pragma unroll
    for (int s = 0; s < 4; ++s) {
      int bn  = 16 * s + lm;
      int bkb = half ? 16 : 0;
      bf4[s] = cat8(ld8(Bt + bn * 32 + bkb), ld8(Bt + bn * 32 + bkb + 8));
    }
#pragma unroll
    for (int s = 0; s < 4; ++s) acc[s] = wmma_bf16(af, bf4[s], acc[s]);
  }

#pragma unroll
  for (int s = 0; s < 4; ++s) {
    int c = n0 + 16 * s + lm;
#pragma unroll
    for (int i = 0; i < 8; ++i) {
      int r = m0 + wave * 16 + i + half * 8;
      float val = acc[s][i];
      if (mode == 0) {
        Cout[(size_t)r * N + c] = val;
      } else {
        int which = c >> 10;          // 0=q 1=k 2=v
        int a = c & 1023;
        int h = a >> 6, d = a & 63;
        int b = r >> 11, l = r & 2047;
        if (which == 2) {             // v stored transposed: (B,H,D,L)
          vto[(((size_t)(b * HH + h)) * DD + d) * LLN + l] = (__bf16)val;
        } else {
          __bf16* dst = (which == 0) ? qo : ko;
          dst[(((size_t)(b * HH + h)) * LLN + l) * DD + d] = (__bf16)val;
        }
      }
    }
  }
}

// ---------------- RoPE (interleaved pairs) in-place on q and k ----------------
__global__ void rope_kernel(__bf16* __restrict__ q, __bf16* __restrict__ k) {
  const int NQ = BB * HH * LLN * (DD / 2);
  int i = blockIdx.x * blockDim.x + threadIdx.x;
  __bf16* tp = (i < NQ) ? q : k;
  int j  = (i < NQ) ? i : i - NQ;
  int dp = j & 31;
  int l  = (j >> 5) & (LLN - 1);
  int bh = j >> 16;
  size_t base = (((size_t)bh) * LLN + l) * DD + 2 * dp;
  float e = (float)tp[base], o = (float)tp[base + 1];
  float ang = (float)l * __powf(10000.0f, -((float)(2 * dp) / 64.0f));
  float sn, cs;
  __sincosf(ang, &sn, &cs);
  tp[base]     = (__bf16)(e * cs - o * sn);
  tp[base + 1] = (__bf16)(o * cs + e * sn);
}

// ---------------- flash attention: 64 q-rows per block, 4 waves ----------------
// v supplied pre-transposed as (B,H,D,L)
__global__ __launch_bounds__(128) void flash_attn(
    const __bf16* __restrict__ q, const __bf16* __restrict__ k,
    const __bf16* __restrict__ vT, __bf16* __restrict__ obf) {
  __shared__ __bf16 Kt[64 * 64];        // (key, d)
  __shared__ __bf16 Vt[64 * 64];        // (d, key)
  __shared__ __bf16 Pst[4 * 16 * 64];   // per-wave P staging

  const int t    = threadIdx.x;
  const int lane = t & 31;
  const int wave = t >> 5;
  const int half = lane >> 4;
  const int lm   = lane & 15;
  const int bh   = blockIdx.y;
  const int b    = bh >> 4, h = bh & 15;
  const int q0   = blockIdx.x * 64;

  const __bf16* qp  = q  + ((size_t)bh * LLN) * DD;
  const __bf16* kp  = k  + ((size_t)bh * LLN) * DD;
  const __bf16* vtp = vT + ((size_t)bh * DD) * LLN;

  v16bf qf[2];
  {
    int row = q0 + wave * 16 + lm;
    const __bf16* rp = qp + (size_t)row * DD;
#pragma unroll
    for (int kk = 0; kk < 2; ++kk) {
      int base = kk * 32 + half * 16;
      qf[kk] = cat8(ld8(rp + base), ld8(rp + base + 8));
    }
  }

  v8f oacc[4] = {};
  float mst[8], lst[8];
#pragma unroll
  for (int i = 0; i < 8; ++i) { mst[i] = -3.0e38f; lst[i] = 0.0f; }

  const float scale = 0.125f;   // 1/sqrt(64)

  for (int kt = 0; kt < LLN / 64; ++kt) {
    __syncthreads();
    // cooperative staging: both tiles are contiguous b128 copies now
    {
      int r = t >> 1, seg = t & 1;
      const __bf16* ks = kp + (size_t)(kt * 64 + r) * DD + seg * 32;
      v8bf a0 = ld8(ks), a1 = ld8(ks + 8), a2 = ld8(ks + 16), a3 = ld8(ks + 24);
      *(v8bf*)(Kt + r * 64 + seg * 32)      = a0;
      *(v8bf*)(Kt + r * 64 + seg * 32 + 8)  = a1;
      *(v8bf*)(Kt + r * 64 + seg * 32 + 16) = a2;
      *(v8bf*)(Kt + r * 64 + seg * 32 + 24) = a3;
      const __bf16* vs = vtp + (size_t)r * LLN + kt * 64 + seg * 32;
      v8bf b0 = ld8(vs), b1 = ld8(vs + 8), b2 = ld8(vs + 16), b3 = ld8(vs + 24);
      *(v8bf*)(Vt + r * 64 + seg * 32)      = b0;
      *(v8bf*)(Vt + r * 64 + seg * 32 + 8)  = b1;
      *(v8bf*)(Vt + r * 64 + seg * 32 + 16) = b2;
      *(v8bf*)(Vt + r * 64 + seg * 32 + 24) = b3;
    }
    __syncthreads();

    // S = Q * K^T   (16 x 64 per wave)
    v8f sacc[4] = {};
#pragma unroll
    for (int kk = 0; kk < 2; ++kk) {
      v16bf kf[4];
#pragma unroll
      for (int s = 0; s < 4; ++s) {
        int brow = 16 * s + lm;
        int db   = kk * 32 + half * 16;
        kf[s] = cat8(ld8(Kt + brow * 64 + db), ld8(Kt + brow * 64 + db + 8));
      }
#pragma unroll
      for (int s = 0; s < 4; ++s) sacc[s] = wmma_bf16(qf[kk], kf[s], sacc[s]);
    }

    // online softmax
#pragma unroll
    for (int i = 0; i < 8; ++i) {
      float v0 = sacc[0][i] * scale, v1 = sacc[1][i] * scale;
      float v2 = sacc[2][i] * scale, v3 = sacc[3][i] * scale;
      float mloc = fmaxf(fmaxf(v0, v1), fmaxf(v2, v3));
#pragma unroll
      for (int off = 1; off < 16; off <<= 1)
        mloc = fmaxf(mloc, __shfl_xor(mloc, off, 32));
      float mnew = fmaxf(mst[i], mloc);
      float corr = __expf(mst[i] - mnew);
      v0 = __expf(v0 - mnew); v1 = __expf(v1 - mnew);
      v2 = __expf(v2 - mnew); v3 = __expf(v3 - mnew);
      float rsum = v0 + v1 + v2 + v3;
#pragma unroll
      for (int off = 1; off < 16; off <<= 1)
        rsum += __shfl_xor(rsum, off, 32);
      lst[i] = lst[i] * corr + rsum;
      mst[i] = mnew;
      oacc[0][i] *= corr; oacc[1][i] *= corr;
      oacc[2][i] *= corr; oacc[3][i] *= corr;
      sacc[0][i] = v0; sacc[1][i] = v1; sacc[2][i] = v2; sacc[3][i] = v3;
    }

    // C-layout P -> A-layout via per-wave LDS staging
    __bf16* pw = Pst + wave * 16 * 64;
#pragma unroll
    for (int s = 0; s < 4; ++s) {
#pragma unroll
      for (int i = 0; i < 8; ++i)
        pw[(i + half * 8) * 64 + 16 * s + lm] = (__bf16)sacc[s][i];
    }
    asm volatile("s_wait_dscnt 0" ::: "memory");

    // O += P * V
#pragma unroll
    for (int kk = 0; kk < 2; ++kk) {
      int pb = kk * 32 + half * 16;
      v16bf pf = cat8(ld8(pw + lm * 64 + pb), ld8(pw + lm * 64 + pb + 8));
      v16bf vf[4];
#pragma unroll
      for (int s = 0; s < 4; ++s) {
        int vd = 16 * s + lm;
        int kb = kk * 32 + half * 16;
        vf[s] = cat8(ld8(Vt + vd * 64 + kb), ld8(Vt + vd * 64 + kb + 8));
      }
#pragma unroll
      for (int s = 0; s < 4; ++s) oacc[s] = wmma_bf16(pf, vf[s], oacc[s]);
    }
  }

  // epilogue: normalize and write O as (B, L, H*D) bf16
#pragma unroll
  for (int s = 0; s < 4; ++s) {
#pragma unroll
    for (int i = 0; i < 8; ++i) {
      int row = q0 + wave * 16 + i + half * 8;
      int d   = 16 * s + lm;
      float val = oacc[s][i] / lst[i];
      obf[((size_t)b * LLN + row) * AAH + h * DD + d] = (__bf16)val;
    }
  }
}

// ---------------- launch ----------------
extern "C" void kernel_launch(void* const* d_in, const int* in_sizes, int n_in,
                              void* d_out, int out_size, void* d_ws, size_t ws_size,
                              hipStream_t stream) {
  const float* x     = (const float*)d_in[0];
  const float* w_qkv = (const float*)d_in[1];
  const float* w_out = (const float*)d_in[2];
  float* out = (float*)d_out;

  char* ws = (char*)d_ws;
  size_t off = 0;
  auto alloc = [&](size_t bytes) -> char* {
    char* p = ws + off;
    off += (bytes + 255) & ~(size_t)255;
    return p;
  };
  __bf16* xbf  = (__bf16*)alloc((size_t)BB * LLN * EE * 2);
  __bf16* wqT  = (__bf16*)alloc((size_t)EE * N3A * 2);        // (3A, E)
  __bf16* woT  = (__bf16*)alloc((size_t)AAH * EE * 2);        // (E, A)
  __bf16* qr   = (__bf16*)alloc((size_t)BB * HH * LLN * DD * 2);
  __bf16* kr   = (__bf16*)alloc((size_t)BB * HH * LLN * DD * 2);
  __bf16* vtr  = (__bf16*)alloc((size_t)BB * HH * DD * LLN * 2);  // (B,H,D,L)
  __bf16* obf  = (__bf16*)alloc((size_t)BB * LLN * AAH * 2);

  cvt_f32_bf16<<<1024, 256, 0, stream>>>(x, xbf, BB * LLN * EE);
  {
    dim3 g(N3A / 64, EE / 64);
    cvt_transpose_bf16<<<g, 256, 0, stream>>>(w_qkv, wqT, EE, N3A);
  }
  {
    dim3 g(EE / 64, AAH / 64);
    cvt_transpose_bf16<<<g, 256, 0, stream>>>(w_out, woT, AAH, EE);
  }

  // QKV projection with scatter epilogue (v written transposed)
  dim3 g1(N3A / 64, (BB * LLN) / 64);
  gemm_bf16_64x64<<<g1, 128, 0, stream>>>(xbf, wqT, BB * LLN, N3A, EE, 1,
                                          nullptr, qr, kr, vtr);

  // RoPE on q and k
  int nrope = 2 * BB * HH * LLN * (DD / 2);
  rope_kernel<<<nrope / 256, 256, 0, stream>>>(qr, kr);

  // flash attention
  dim3 g2(LLN / 64, BB * HH);
  flash_attn<<<g2, 128, 0, stream>>>(qr, kr, vtr, obf);

  // output projection -> fp32 d_out
  dim3 g3(EE / 64, (BB * LLN) / 64);
  gemm_bf16_64x64<<<g3, 128, 0, stream>>>(obf, woT, BB * LLN, EE, AAH, 0,
                                          out, nullptr, nullptr, nullptr);
}
